// OT_71073118814560
// MI455X (gfx1250) — compile-verified, matrix operand order checked by reading, not run on hardware
//
#include <hip/hip_runtime.h>
#include <stdint.h>

// ---------------------------------------------------------------------------
// CDNA5 (gfx1250) implementation of the OT loss pipeline.
// bf16 WMMA (v_wmma_f32_16x16x32_bf16) for all GEMMs, fp32 everywhere else.
// ---------------------------------------------------------------------------

typedef __attribute__((ext_vector_type(16))) __bf16 v16bf;
typedef __attribute__((ext_vector_type(8)))  float  v8f;

constexpr int kBS   = 64;
constexpr int kN    = 512;
constexpr int kD    = 256;
constexpr int kRows = kBS * kN;     // 32768 rows per tensor

union BF16Frag { v16bf v; unsigned short s[16]; };

__device__ __forceinline__ unsigned short f2bf(float f) {
  unsigned int u = __float_as_uint(f);
  u += 0x7FFFu + ((u >> 16) & 1u);            // round-to-nearest-even
  return (unsigned short)(u >> 16);
}
__device__ __forceinline__ float bf2f(unsigned short s) {
  return __uint_as_float(((unsigned int)s) << 16);
}

// A/B fragment loader, 16x32 bf16, from an fp32 row (stride kD floats).
// lane&15 = row/col index within tile, lane>>4 = khalf (ISA 7.12.2 layout).
__device__ __forceinline__ v16bf load_frag_f32(const float* row, int kc, int khalf) {
  BF16Frag f;
  const int k0 = kc * 32 + khalf * 8;
#pragma unroll
  for (int t = 0; t < 8; ++t) f.s[t]     = f2bf(row[k0 + t]);
#pragma unroll
  for (int t = 0; t < 8; ++t) f.s[8 + t] = f2bf(row[k0 + 16 + t]);
  return f.v;
}
__device__ __forceinline__ v16bf load_frag_bf16(const unsigned short* row, int kc, int khalf) {
  BF16Frag f;
  const int k0 = kc * 32 + khalf * 8;
#pragma unroll
  for (int t = 0; t < 8; ++t) f.s[t]     = row[k0 + t];
#pragma unroll
  for (int t = 0; t < 8; ++t) f.s[8 + t] = row[k0 + 16 + t];
  return f.v;
}

__device__ __forceinline__ v8f wmma_bf16(v16bf a, v16bf b, v8f c) {
  // 8 args: (neg_a, A, neg_b, B, c_mod, C, reuse_a, reuse_b)
  return __builtin_amdgcn_wmma_f32_16x16x32_bf16(false, a, false, b, (short)0, c,
                                                 false, false);
}

// ---------------------------------------------------------------------------
// Small prep kernels
// ---------------------------------------------------------------------------

__global__ void init_accum_kernel(float* accum) {
  if (threadIdx.x < 4) accum[threadIdx.x] = 0.0f;
}

// Wt[n][k] = bf16(W[k][n]) so that WMMA B-operand (column-major) loads are rows.
__global__ void prep_weights_kernel(const float* __restrict__ W1,
                                    const float* __restrict__ W2,
                                    unsigned short* __restrict__ W1t,
                                    unsigned short* __restrict__ W2t) {
  int t = blockIdx.x * blockDim.x + threadIdx.x;   // 65536
  int n = t & 255, k = t >> 8;
  W1t[(size_t)n * kD + k] = f2bf(W1[(size_t)k * kD + n]);
  W2t[(size_t)n * kD + k] = f2bf(W2[(size_t)k * kD + n]);
}

// argmax over 4 logits (first-max tie-break, matching jnp.argmax)
__global__ void argmax4_kernel(const float* __restrict__ la,
                               const float* __restrict__ lb,
                               int* __restrict__ argA, int* __restrict__ argB) {
  int t = blockIdx.x * blockDim.x + threadIdx.x;   // 65536
  const float* l = (t < kRows) ? la + (size_t)t * 4 : lb + (size_t)(t - kRows) * 4;
  int* dst       = (t < kRows) ? argA + t          : argB + (t - kRows);
  float best = l[0]; int bi = 0;
#pragma unroll
  for (int c = 1; c < 4; ++c) { float v = l[c]; if (v > best) { best = v; bi = c; } }
  *dst = bi;
}

// ---------------------------------------------------------------------------
// Embed: h = LN(lrelu(x@W1+b1)); e = LN(lrelu(h@W2+b2)); store bf16 + sqnorm.
// 1 wave per 16-row tile, 4 waves / block, 16KB LDS per wave.
// ---------------------------------------------------------------------------

__device__ __forceinline__ void layernorm16(float (*rows)[kD],
                                            const float* __restrict__ g,
                                            const float* __restrict__ be,
                                            int lane) {
  const int row = lane & 15;
  const int c0  = (lane >> 4) * 128;     // two lanes cooperate per row
  float s = 0.0f;
  for (int c = 0; c < 128; ++c) s += rows[row][c0 + c];
  s += __shfl_xor(s, 16);
  const float mean = s * (1.0f / 256.0f);
  float v = 0.0f;
  for (int c = 0; c < 128; ++c) { float d = rows[row][c0 + c] - mean; v += d * d; }
  v += __shfl_xor(v, 16);
  const float rs = rsqrtf(v * (1.0f / 256.0f) + 1e-5f);
  for (int c = 0; c < 128; ++c) {
    const int col = c0 + c;
    rows[row][col] = (rows[row][col] - mean) * rs * g[col] + be[col];
  }
}

__global__ void embed_kernel(const float* __restrict__ X,
                             const unsigned short* __restrict__ W1t,
                             const float* __restrict__ b1,
                             const float* __restrict__ g1,
                             const float* __restrict__ be1,
                             const unsigned short* __restrict__ W2t,
                             const float* __restrict__ b2,
                             const float* __restrict__ g2,
                             const float* __restrict__ be2,
                             unsigned short* __restrict__ E,
                             float* __restrict__ Nrm) {
  __shared__ float smem[4][16][kD];                // 64 KB
  const int lane  = threadIdx.x & 31;
  const int wv    = threadIdx.x >> 5;
  const int r0    = (blockIdx.x * 4 + wv) * 16;    // exact grid: no guard needed
  const int nlo   = lane & 15;
  const int khalf = lane >> 4;

  v16bf afr[8];
  const float* xrow = X + (size_t)(r0 + nlo) * kD;
#pragma unroll
  for (int kc = 0; kc < 8; ++kc) afr[kc] = load_frag_f32(xrow, kc, khalf);

  // ---- layer 1: x@W1 + b1, leaky relu ----
  for (int nc = 0; nc < 16; ++nc) {
    v8f acc = {};
#pragma unroll
    for (int kc = 0; kc < 8; ++kc) {
      v16bf bfr = load_frag_bf16(W1t + (size_t)(nc * 16 + nlo) * kD, kc, khalf);
      acc = wmma_bf16(afr[kc], bfr, acc);
    }
    const int col = nc * 16 + nlo;
    const float bias = b1[col];
#pragma unroll
    for (int v = 0; v < 8; ++v) {
      const int m = khalf * 8 + v;                 // C/D layout: row = 8*(lane>>4)+v
      float val = acc[v] + bias;
      val = val > 0.0f ? val : 0.01f * val;
      smem[wv][m][col] = val;
    }
  }
  __syncthreads();
  layernorm16(smem[wv], g1, be1, lane);
  __syncthreads();

  // ---- re-fragment h from LDS ----
#pragma unroll
  for (int kc = 0; kc < 8; ++kc) afr[kc] = load_frag_f32(&smem[wv][nlo][0], kc, khalf);
  __syncthreads();

  // ---- layer 2: h@W2 + b2, leaky relu ----
  for (int nc = 0; nc < 16; ++nc) {
    v8f acc = {};
#pragma unroll
    for (int kc = 0; kc < 8; ++kc) {
      v16bf bfr = load_frag_bf16(W2t + (size_t)(nc * 16 + nlo) * kD, kc, khalf);
      acc = wmma_bf16(afr[kc], bfr, acc);
    }
    const int col = nc * 16 + nlo;
    const float bias = b2[col];
#pragma unroll
    for (int v = 0; v < 8; ++v) {
      const int m = khalf * 8 + v;
      float val = acc[v] + bias;
      val = val > 0.0f ? val : 0.01f * val;
      smem[wv][m][col] = val;
    }
  }
  __syncthreads();
  layernorm16(smem[wv], g2, be2, lane);
  __syncthreads();

  // ---- write bf16 embedding + squared norm (of the bf16-rounded values) ----
  {
    const int row = nlo;
    const int c0  = khalf * 128;
    unsigned short* erow = E + (size_t)(r0 + row) * kD;
    float ns = 0.0f;
    for (int c = 0; c < 128; ++c) {
      const unsigned short bv = f2bf(smem[wv][row][c0 + c]);
      erow[c0 + c] = bv;
      const float fv = bf2f(bv);
      ns += fv * fv;
    }
    ns += __shfl_xor(ns, 16);
    if (khalf == 0) Nrm[r0 + row] = ns;
  }
}

// ---------------------------------------------------------------------------
// Cost: G = eA·eB^T (WMMA), cost_class = sqrt(max(na+nb-2G,0)) + BIG*(1-pm).
// Accumulate in/out loss partials; store K = exp(-cost_class/REG) for Sinkhorn.
// 1 wave per 16x16 tile, 8 waves / block.
// ---------------------------------------------------------------------------

__global__ void cost_kernel(const unsigned short* __restrict__ eA,
                            const unsigned short* __restrict__ eB,
                            const float* __restrict__ na,
                            const float* __restrict__ nb,
                            const int* __restrict__ argA,
                            const int* __restrict__ argB,
                            const float* __restrict__ pma,
                            const float* __restrict__ pmb,
                            float* __restrict__ Kmat,
                            float* __restrict__ accum) {
  const int lane  = threadIdx.x & 31;
  const int wv    = threadIdx.x >> 5;
  const int tile  = blockIdx.x * 8 + wv;           // 0..65535
  const int b     = tile >> 10;
  const int t     = tile & 1023;
  const int ti    = t >> 5;
  const int tj    = t & 31;
  const int nlo   = lane & 15;
  const int khalf = lane >> 4;

  const unsigned short* arow = eA + ((size_t)b * kN + ti * 16 + nlo) * kD;
  const unsigned short* brow = eB + ((size_t)b * kN + tj * 16 + nlo) * kD;

  v8f acc = {};
#pragma unroll
  for (int kc = 0; kc < 8; ++kc) {
    v16bf af = load_frag_bf16(arow, kc, khalf);
    v16bf bf = load_frag_bf16(brow, kc, khalf);   // B operand = eB rows (column-major B)
    acc = wmma_bf16(af, bf, acc);
  }

  const int j = tj * 16 + nlo;                     // C/D layout: col = lane&15
  const size_t bj = (size_t)b * kN + j;
  const float nbj = nb[bj];
  const float pbj = pmb[bj];
  const int   abj = argB[bj];

  float in_p = 0.0f, out_p = 0.0f;
#pragma unroll
  for (int v = 0; v < 8; ++v) {
    const int i = ti * 16 + khalf * 8 + v;         // row = 8*(lane>>4)+v
    const size_t bi = (size_t)b * kN + i;
    const float sq   = na[bi] + nbj - 2.0f * acc[v];
    const float cost = sqrtf(fmaxf(sq, 0.0f));
    const float pm   = pma[bi] * pbj;
    const float cc   = cost + 1.0e9f * (1.0f - pm);
    const float mask = (argA[bi] == abj) ? 1.0f : 0.0f;
    in_p  += cc * mask * pm;
    out_p += cc * (1.0f - mask) * pm;
    Kmat[bi * kN + j] = expf(-2.0f * cc);          // 1/REG = 2
  }
  // wave reduction -> single atomic per wave
#pragma unroll
  for (int off = 16; off > 0; off >>= 1) {
    in_p  += __shfl_down(in_p,  off);
    out_p += __shfl_down(out_p, off);
  }
  if (lane == 0) {
    atomicAdd(&accum[0], in_p);
    atomicAdd(&accum[1], out_p);
  }
}

// ---------------------------------------------------------------------------
// Sinkhorn: 1 block (512 threads) per batch; u,v in LDS; K resident in L2.
// Then sup-loss partial from plan = u .* K .* v.
// ---------------------------------------------------------------------------

__global__ void sinkhorn_kernel(const float* __restrict__ Kmat,
                                const float* __restrict__ rel,
                                const float* __restrict__ pma,
                                float* __restrict__ accum) {
  __shared__ float su[kN];
  __shared__ float sv[kN];
  const int b = blockIdx.x;
  const int t = threadIdx.x;
  const float* Kb = Kmat + (size_t)b * kN * kN;
  const float m = 1.0f / (float)kN;

  su[t] = m;
  sv[t] = m;
  __syncthreads();

  for (int it = 0; it < 10; ++it) {
    // v = margin / (K^T u)   (coalesced: row i contiguous across threads t=j)
    float s = 0.0f;
    for (int i = 0; i < kN; ++i) s += Kb[(size_t)i * kN + t] * su[i];
    sv[t] = m / s;
    __syncthreads();
    // u = margin / (K v)
    float s2 = 0.0f;
    const float* row = Kb + (size_t)t * kN;
    for (int j = 0; j < kN; ++j) s2 += row[j] * sv[j];
    su[t] = m / s2;
    __syncthreads();
  }

  // sup loss partial: sum_j (plan*n - rel)^2 * pma[b,i]
  const float ui = su[t];
  const float pa = pma[(size_t)b * kN + t];
  const float* row  = Kb  + (size_t)t * kN;
  const float* rrow = rel + ((size_t)b * kN + t) * kN;
  float acc = 0.0f;
  for (int j = 0; j < kN; ++j) {
    const float plan = ui * row[j] * sv[j];
    const float d = plan * (float)kN - rrow[j];
    acc += d * d;
  }
  atomicAdd(&accum[2], acc * pa);
}

// ---------------------------------------------------------------------------
// Finalize: out = (in_sum - out_sum)/ (BS*N*N) + 10 * sup_sum / sum(pma)
// ---------------------------------------------------------------------------

__global__ void finalize_kernel(const float* __restrict__ pma,
                                const float* __restrict__ accum,
                                float* __restrict__ out) {
  __shared__ float red[256];
  const int t = threadIdx.x;
  float s = 0.0f;
  for (int i = t; i < kRows; i += 256) s += pma[i];
  red[t] = s;
  __syncthreads();
  for (int w = 128; w > 0; w >>= 1) {
    if (t < w) red[t] += red[t + w];
    __syncthreads();
  }
  if (t == 0) {
    const float invM = 1.0f / ((float)kBS * (float)kN * (float)kN);
    out[0] = accum[0] * invM - accum[1] * invM + 10.0f * accum[2] / red[0];
  }
}

// ---------------------------------------------------------------------------
// Launch
// ---------------------------------------------------------------------------

extern "C" void kernel_launch(void* const* d_in, const int* in_sizes, int n_in,
                              void* d_out, int out_size, void* d_ws, size_t ws_size,
                              hipStream_t stream) {
  (void)in_sizes; (void)n_in; (void)out_size; (void)ws_size;

  const float* Alog = (const float*)d_in[0];
  const float* Blog = (const float*)d_in[1];
  const float* Ax   = (const float*)d_in[2];
  const float* Bx   = (const float*)d_in[3];
  const float* rel  = (const float*)d_in[4];
  const float* pma  = (const float*)d_in[5];
  const float* pmb  = (const float*)d_in[6];
  const float* W1   = (const float*)d_in[7];
  const float* b1   = (const float*)d_in[8];
  const float* g1   = (const float*)d_in[9];
  const float* be1  = (const float*)d_in[10];
  const float* W2   = (const float*)d_in[11];
  const float* b2   = (const float*)d_in[12];
  const float* g2   = (const float*)d_in[13];
  const float* be2  = (const float*)d_in[14];
  float* out = (float*)d_out;

  // workspace layout (bytes), ~97 MB total
  char* w = (char*)d_ws;
  unsigned short* eA   = (unsigned short*)(w + 0);          // 32768*256*2 = 16 MB
  unsigned short* eB   = (unsigned short*)(w + 16777216);   // 16 MB
  float*          na   = (float*)(w + 33554432);            // 128 KB
  float*          nb   = (float*)(w + 33685504);            // 128 KB
  int*            argA = (int*)  (w + 33816576);            // 128 KB
  int*            argB = (int*)  (w + 33947648);            // 128 KB
  unsigned short* W1t  = (unsigned short*)(w + 34078720);   // 128 KB
  unsigned short* W2t  = (unsigned short*)(w + 34209792);   // 128 KB
  float*          accum= (float*)(w + 34340864);            // 256 B
  float*          Kmat = (float*)(w + 34341120);            // 64*512*512*4 = 64 MB

  init_accum_kernel<<<1, 32, 0, stream>>>(accum);
  prep_weights_kernel<<<256, 256, 0, stream>>>(W1, W2, W1t, W2t);
  argmax4_kernel<<<256, 256, 0, stream>>>(Alog, Blog, argA, argB);

  // 32768 rows -> 2048 tiles of 16 rows -> 512 blocks * 4 waves
  embed_kernel<<<512, 128, 0, stream>>>(Ax, W1t, b1, g1, be1, W2t, b2, g2, be2, eA, na);
  embed_kernel<<<512, 128, 0, stream>>>(Bx, W1t, b1, g1, be1, W2t, b2, g2, be2, eB, nb);

  // 64 batches * 32*32 tiles = 65536 tiles -> 8192 blocks * 8 waves
  cost_kernel<<<8192, 256, 0, stream>>>(eA, eB, na, nb, argA, argB, pma, pmb, Kmat, accum);

  sinkhorn_kernel<<<kBS, kN, 0, stream>>>(Kmat, rel, pma, accum);
  finalize_kernel<<<1, 256, 0, stream>>>(pma, accum, out);
}
